// TransformerEncoderLayer_26362509263198
// MI455X (gfx1250) — matrix operand order repaired and checked by reading, not compiled, not run on hardware
//
#include <hip/hip_runtime.h>
#include <hip/hip_bf16.h>
#include <cmath>

#define DMODEL 1024
#define NHEADS 16
#define DK     64
#define DFF    4096
#define DSTATE 16
#define DCONV  4
#define DINNER 2048
#define NBATCH 2
#define SEQ    2048
#define MTOK   (NBATCH * SEQ)   // 4096 tokens
#define LN_EPS 1e-5f

typedef __attribute__((ext_vector_type(16))) __bf16 bf16x16;
typedef __attribute__((ext_vector_type(8)))  __bf16 bf16x8;
typedef __attribute__((ext_vector_type(8)))  float  f32x8;
typedef __attribute__((ext_vector_type(4)))  unsigned uint32x4;
typedef __attribute__((ext_vector_type(8)))  unsigned uint32x8;

// ---------------------------------------------------------------- helpers ---
__device__ __forceinline__ f32x8 wmma_bf16(bf16x16 a, bf16x16 b, f32x8 c) {
  // (neg_a, A, neg_b, B, c_mod, C, reuse_a, reuse_b)
  return __builtin_amdgcn_wmma_f32_16x16x32_bf16(false, a, false, b, (short)0, c,
                                                 false, false);
}

__device__ __forceinline__ bf16x16 pack16(bf16x8 lo, bf16x8 hi) {
  bf16x16 r;
#pragma unroll
  for (int i = 0; i < 8; ++i) { r[i] = lo[i]; r[i + 8] = hi[i]; }
  return r;
}

__device__ __forceinline__ float red_max16(float x) {
#pragma unroll
  for (int m = 8; m >= 1; m >>= 1) x = fmaxf(x, __shfl_xor(x, m, 32));
  return x;
}
__device__ __forceinline__ float red_sum16(float x) {
#pragma unroll
  for (int m = 8; m >= 1; m >>= 1) x += __shfl_xor(x, m, 32);
  return x;
}
__device__ __forceinline__ float gelu_tanh(float x) {
  return 0.5f * x * (1.f + tanhf(0.7978845608028654f * (x + 0.044715f * x * x * x)));
}

// Tensor Data Mover: DMA a [64 rows x 32 cols] bf16 tile of a row-major
// (rows x K) matrix starting at gsrc into LDS at lds_off (contiguous).
// 2D descriptor: groups 0 and 1 only (VADDR2/3 omitted -> NULL).
__device__ __forceinline__ void tdm_load_tile_64x32(const __bf16* gsrc,
                                                    unsigned lds_off,
                                                    int K, int Nrows) {
  unsigned long long ga = (unsigned long long)(size_t)gsrc;
  uint32x4 g0;
  uint32x8 g1;
  g0[0] = 1u;                                            // count=1, user desc
  g0[1] = lds_off;                                       // lds_addr
  g0[2] = (unsigned)(ga & 0xffffffffu);                  // global_addr[31:0]
  g0[3] = (unsigned)((ga >> 32) & 0x01ffffffu)           // global_addr[56:32]
          | (2u << 30);                                  // type=2 (image)
  g1[0] = 1u << 16;                                      // data_size=1 (2 bytes)
  g1[1] = ((unsigned)K & 0xffffu) << 16;                 // tensor_dim0[15:0]
  g1[2] = (((unsigned)K >> 16) & 0xffffu)                // tensor_dim0[31:16]
          | (((unsigned)Nrows & 0xffffu) << 16);         // tensor_dim1[15:0]
  g1[3] = (((unsigned)Nrows >> 16) & 0xffffu)            // tensor_dim1[31:16]
          | (32u << 16);                                 // tile_dim0 = 32
  g1[4] = 64u;                                           // tile_dim1 = 64
  g1[5] = (unsigned)K;                                   // dim0_stride[31:0]
  g1[6] = 0u;                                            // dim0_stride[47:32], dim1_stride lo
  g1[7] = 0u;
  asm volatile("tensor_load_to_lds %0, %1" : : "s"(g0), "s"(g1) : "memory");
}

// ------------------------------------------------------------ cvt kernel ---
__global__ void cvt_f32_to_bf16(const float* __restrict__ src,
                                __bf16* __restrict__ dst, int n) {
  int i = blockIdx.x * blockDim.x + threadIdx.x;
  if (i < n) dst[i] = (__bf16)src[i];
}

// ------------------------------------------------- WMMA GEMM (TDM-staged) --
// C[M,N] = A[M,K] * W[N,K]^T + bias ; A,W bf16 row-major (K contiguous).
// block = 128 threads (4 waves); wave w -> M-tile blockIdx.y*4+w; the block
// shares one 64-column N strip. W k-slabs are DMA'd into LDS by the TDM
// (double buffered), B-fragments are read from LDS; A streams from L2.
__global__ void gemm_bf16_tdm(const __bf16* __restrict__ A,
                              const __bf16* __restrict__ W,
                              const float* __restrict__ bias,
                              float* __restrict__ C,
                              __bf16* __restrict__ C16,
                              int M, int N, int K, int act) {
  __shared__ __align__(16) __bf16 wbuf[2][64][32];
  const int lane = threadIdx.x & 31;
  const int wave = threadIdx.x >> 5;
  const int l16  = lane & 15;
  const int half = lane >> 4;
  const int mt   = blockIdx.y * 4 + wave;
  const int n0   = blockIdx.x * 64;

  const size_t arow = (size_t)(mt * 16 + l16) * K;

  f32x8 acc[4];
#pragma unroll
  for (int t = 0; t < 4; ++t) acc[t] = {};

  if (wave == 0) {
    tdm_load_tile_64x32(W + (size_t)n0 * K, (unsigned)(size_t)&wbuf[0][0][0], K, N);
    __builtin_amdgcn_s_wait_tensorcnt(0);
  }
  __syncthreads();

  int cur = 0;
  for (int k0 = 0; k0 < K; k0 += 32) {
    if (wave == 0 && k0 + 32 < K)
      tdm_load_tile_64x32(W + (size_t)n0 * K + k0 + 32,
                          (unsigned)(size_t)&wbuf[cur ^ 1][0][0], K, N);
    if (k0 + 128 < K) __builtin_prefetch(A + arow + k0 + 128, 0, 0);

    bf16x16 av = pack16(*(const bf16x8*)(A + arow + k0 + half * 8),
                        *(const bf16x8*)(A + arow + k0 + 16 + half * 8));
#pragma unroll
    for (int t = 0; t < 4; ++t) {
      bf16x16 bv = pack16(*(const bf16x8*)&wbuf[cur][t * 16 + l16][half * 8],
                          *(const bf16x8*)&wbuf[cur][t * 16 + l16][16 + half * 8]);
      acc[t] = wmma_bf16(av, bv, acc[t]);
    }

    if (wave == 0) __builtin_amdgcn_s_wait_tensorcnt(0);
    __syncthreads();
    cur ^= 1;
  }

#pragma unroll
  for (int t = 0; t < 4; ++t) {
    const int ncol = n0 + t * 16 + l16;
    const float bn = bias ? bias[ncol] : 0.f;
#pragma unroll
    for (int v = 0; v < 8; ++v) {
      const int m = mt * 16 + half * 8 + v;
      float val = acc[t][v] + bn;
      if (act == 1) val = gelu_tanh(val);
      const size_t idx = (size_t)m * N + ncol;
      C[idx] = val;
      if (C16) C16[idx] = (__bf16)val;
    }
  }
}

// ------------------------------------------- WMMA GEMM (direct, small N) ---
template <int NT>
__global__ void gemm_bf16_nt(const __bf16* __restrict__ A,
                             const __bf16* __restrict__ W,
                             const float* __restrict__ bias,
                             float* __restrict__ C,
                             __bf16* __restrict__ C16,
                             int M, int N, int K, int act) {
  const int lane = threadIdx.x & 31;
  const int wave = threadIdx.x >> 5;
  const int l16  = lane & 15;
  const int half = lane >> 4;
  const int mt   = blockIdx.y * 4 + wave;
  const int n0   = blockIdx.x * 16 * NT;
  if (mt * 16 >= M) return;

  const size_t arow = (size_t)(mt * 16 + l16) * K;

  f32x8 acc[NT];
#pragma unroll
  for (int t = 0; t < NT; ++t) acc[t] = {};

  for (int k0 = 0; k0 < K; k0 += 32) {
    if (k0 + 128 < K) __builtin_prefetch(A + arow + k0 + 128, 0, 0);
    bf16x16 av = pack16(*(const bf16x8*)(A + arow + k0 + half * 8),
                        *(const bf16x8*)(A + arow + k0 + 16 + half * 8));
#pragma unroll
    for (int t = 0; t < NT; ++t) {
      const size_t wrow = (size_t)(n0 + t * 16 + l16) * K;
      bf16x8 blo = *(const bf16x8*)(W + wrow + k0 + half * 8);
      bf16x8 bhi = *(const bf16x8*)(W + wrow + k0 + 16 + half * 8);
      acc[t] = wmma_bf16(av, pack16(blo, bhi), acc[t]);
    }
  }

#pragma unroll
  for (int t = 0; t < NT; ++t) {
    const int ncol = n0 + t * 16 + l16;
    const float bn = bias ? bias[ncol] : 0.f;
#pragma unroll
    for (int v = 0; v < 8; ++v) {
      const int m = mt * 16 + half * 8 + v;
      float val = acc[t][v] + bn;
      if (act == 1) val = gelu_tanh(val);
      const size_t idx = (size_t)m * N + ncol;
      C[idx] = val;
      if (C16) C16[idx] = (__bf16)val;
    }
  }
}

// ---------------------------------------------------- V transpose (attn) ---
// vt16[((b*16+h)*64+d)*2048 + s] = bf16( v[(b*2048+s)*1024 + h*64+d] )
__global__ void transpose_v(const float* __restrict__ v, __bf16* __restrict__ vt) {
  long i = (long)blockIdx.x * blockDim.x + threadIdx.x;  // over MTOK*DMODEL
  if (i >= (long)MTOK * DMODEL) return;
  int col = (int)(i & (DMODEL - 1));
  int t   = (int)(i >> 10);
  int b = t >> 11, s = t & (SEQ - 1);
  int h = col >> 6, d = col & 63;
  vt[(((long)((b * NHEADS + h) * DK + d)) << 11) + s] = (__bf16)v[i];
}

// -------------------------------------------------- flash attention (WMMA) -
// grid (SEQ/64, NHEADS, NBATCH), block 128 (4 waves, 16 queries each).
__global__ void flash_attn(const __bf16* __restrict__ q16,
                           const __bf16* __restrict__ k16,
                           const __bf16* __restrict__ vt16,
                           const int* __restrict__ mask,
                           float* __restrict__ out) {
  __shared__ __align__(16) __bf16 pbuf[4][16][32];  // per-wave P tile
  const int lane = threadIdx.x & 31;
  const int wave = threadIdx.x >> 5;
  const int l16  = lane & 15;
  const int half = lane >> 4;
  const int h = blockIdx.y, b = blockIdx.z;
  const int q0 = (blockIdx.x * 4 + wave) * 16;

  // Q fragments (A-layout, head dim K split 0..31 / 32..63)
  const long qrow = (long)(b * SEQ + q0 + l16) * DMODEL + h * DK;
  bf16x16 aq0 = pack16(*(const bf16x8*)(q16 + qrow + half * 8),
                       *(const bf16x8*)(q16 + qrow + 16 + half * 8));
  bf16x16 aq1 = pack16(*(const bf16x8*)(q16 + qrow + 32 + half * 8),
                       *(const bf16x8*)(q16 + qrow + 48 + half * 8));

  f32x8 acc0 = {}, acc1 = {}, acc2 = {}, acc3 = {};
  float mrun[8], lrun[8];
#pragma unroll
  for (int v = 0; v < 8; ++v) { mrun[v] = -INFINITY; lrun[v] = 0.f; }

  for (int key0 = 0; key0 < SEQ; key0 += 32) {
    // ---- S = Q K^T for two 16-key subtiles
    f32x8 s0 = {}, s1 = {};
#pragma unroll
    for (int t = 0; t < 2; ++t) {
      const long krow = (long)(b * SEQ + key0 + 16 * t + l16) * DMODEL + h * DK;
      bf16x16 bk0 = pack16(*(const bf16x8*)(k16 + krow + half * 8),
                           *(const bf16x8*)(k16 + krow + 16 + half * 8));
      bf16x16 bk1 = pack16(*(const bf16x8*)(k16 + krow + 32 + half * 8),
                           *(const bf16x8*)(k16 + krow + 48 + half * 8));
      f32x8 st = {};
      st = wmma_bf16(aq0, bk0, st);
      st = wmma_bf16(aq1, bk1, st);
      const int mk = mask[b * SEQ + key0 + 16 * t + l16];
#pragma unroll
      for (int v = 0; v < 8; ++v) {
        float x = st[v] * 0.125f;            // 1/sqrt(64)
        st[v] = (mk == 0) ? -INFINITY : x;
      }
      if (t == 0) s0 = st; else s1 = st;
    }

    // ---- online softmax update (rows live across the 16-lane group)
#pragma unroll
    for (int v = 0; v < 8; ++v) {
      float rmax = red_max16(fmaxf(s0[v], s1[v]));
      float nm   = fmaxf(mrun[v], rmax);
      float corr = __expf(mrun[v] - nm);
      mrun[v] = nm;
      float p0 = __expf(s0[v] - nm);
      float p1 = __expf(s1[v] - nm);
      float rs = red_sum16(p0 + p1);
      lrun[v] = lrun[v] * corr + rs;
      s0[v] = p0; s1[v] = p1;
      acc0[v] *= corr; acc1[v] *= corr; acc2[v] *= corr; acc3[v] *= corr;
    }

    // ---- stage P in LDS (C-layout) and reload in A-layout
#pragma unroll
    for (int v = 0; v < 8; ++v) {
      pbuf[wave][half * 8 + v][l16]      = (__bf16)s0[v];
      pbuf[wave][half * 8 + v][16 + l16] = (__bf16)s1[v];
    }
    bf16x16 ap = pack16(*(const bf16x8*)&pbuf[wave][l16][half * 8],
                        *(const bf16x8*)&pbuf[wave][l16][16 + half * 8]);

    // ---- acc += P V  (V pre-transposed: contiguous along keys)
#pragma unroll
    for (int c = 0; c < 4; ++c) {
      const long vrow =
          ((long)((b * NHEADS + h) * DK + c * 16 + l16)) * SEQ + key0 + half * 8;
      bf16x16 bv = pack16(*(const bf16x8*)(vt16 + vrow),
                          *(const bf16x8*)(vt16 + vrow + 16));
      f32x8 a = (c == 0) ? acc0 : (c == 1) ? acc1 : (c == 2) ? acc2 : acc3;
      a = wmma_bf16(ap, bv, a);
      if (c == 0) acc0 = a; else if (c == 1) acc1 = a;
      else if (c == 2) acc2 = a; else acc3 = a;
    }
  }

  // ---- epilogue: divide by l, scatter to [b,s, h*64+d]
#pragma unroll
  for (int c = 0; c < 4; ++c) {
    f32x8 a = (c == 0) ? acc0 : (c == 1) ? acc1 : (c == 2) ? acc2 : acc3;
#pragma unroll
    for (int v = 0; v < 8; ++v) {
      const int m = q0 + half * 8 + v;
      out[(long)(b * SEQ + m) * DMODEL + h * DK + c * 16 + l16] = a[v] / lrun[v];
    }
  }
}

// ------------------------------------------------------- add + LayerNorm ---
// one block (256 thr) per row of 1024
__global__ void add_ln(const float* __restrict__ x, const float* __restrict__ y,
                       const float* __restrict__ g, const float* __restrict__ bt,
                       float* __restrict__ out, __bf16* __restrict__ out16) {
  __shared__ float red[2][8];
  const int row = blockIdx.x;
  const int tid = threadIdx.x;
  const int lane = tid & 31, wave = tid >> 5;
  const float* xr = x + (size_t)row * DMODEL;
  const float* yr = y + (size_t)row * DMODEL;
  float v[4], s = 0.f, s2 = 0.f;
#pragma unroll
  for (int j = 0; j < 4; ++j) {
    float t = xr[tid + j * 256] + yr[tid + j * 256];
    v[j] = t; s += t; s2 += t * t;
  }
#pragma unroll
  for (int o = 16; o >= 1; o >>= 1) { s += __shfl_xor(s, o, 32); s2 += __shfl_xor(s2, o, 32); }
  if (lane == 0) { red[0][wave] = s; red[1][wave] = s2; }
  __syncthreads();
  float ts = 0.f, ts2 = 0.f;
#pragma unroll
  for (int w = 0; w < 8; ++w) { ts += red[0][w]; ts2 += red[1][w]; }
  const float mean = ts * (1.f / DMODEL);
  const float var  = ts2 * (1.f / DMODEL) - mean * mean;
  const float rstd = rsqrtf(var + LN_EPS);
#pragma unroll
  for (int j = 0; j < 4; ++j) {
    const int c = tid + j * 256;
    const float o = (v[j] - mean) * rstd * g[c] + bt[c];
    out[(size_t)row * DMODEL + c] = o;
    if (out16) out16[(size_t)row * DMODEL + c] = (__bf16)o;
  }
}

// --------------------------------------------- depthwise conv + SiLU -------
__global__ void conv_silu(const float* __restrict__ xr, const float* __restrict__ cw,
                          const float* __restrict__ cb, __bf16* __restrict__ o16) {
  long i = (long)blockIdx.x * blockDim.x + threadIdx.x;  // over MTOK*DINNER
  if (i >= (long)MTOK * DINNER) return;
  const int c = (int)(i & (DINNER - 1));
  const long t = i >> 11;
  const int b = (int)(t >> 11), s = (int)(t & (SEQ - 1));
  float acc = cb[c];
#pragma unroll
  for (int j = 0; j < DCONV; ++j) {
    const int sp = s - (DCONV - 1) + j;
    if (sp >= 0) acc += cw[c * DCONV + j] * xr[((long)(b * SEQ + sp)) * (2 * DINNER) + c];
  }
  const float sv = acc * (1.f / (1.f + __expf(-acc)));
  o16[i] = (__bf16)sv;
}

// ------------------------------------------------------ sequential scan ----
// 1 block, 64 threads: wave 0 -> batch 0, wave 1 -> batch 1.
__global__ void mamba_scan(const float* __restrict__ xdbl, const float* __restrict__ dtp,
                           const float* __restrict__ A, float* __restrict__ hs) {
  __shared__ float Aexp[16][16][16];
  const int tid = threadIdx.x;
  for (int idx = tid; idx < 4096; idx += 64) {
    const int i = idx >> 8, sp = (idx >> 4) & 15, t = idx & 15;
    Aexp[i][sp][t] = __expf(__expf(dtp[i]) * A[sp * 16 + t]);
  }
  __syncthreads();
  const int b = tid >> 5;
  const int lane = tid & 31;
  const int j = lane & 15;
  float h = 0.f;
  for (int s = 0; s < SEQ; ++s) {
    const int token = b * SEQ + s;
    const int i = s & 15;
    const float bj = xdbl[token * 32 + j];
    const float cj = xdbl[token * 32 + 16 + j];
    float acc = 0.f;
#pragma unroll
    for (int t = 0; t < 16; ++t) acc += Aexp[i][j][t] * __shfl(h, t, 32);
    h = tanhf(acc + bj * h + cj);
    if (lane < 16) hs[token * 16 + j] = h;
  }
}

// --------------------------------------------- tile(hs) + residual -> bf16 -
__global__ void tile_add_res(const float* __restrict__ hs, const float* __restrict__ xr,
                             __bf16* __restrict__ y16) {
  long i = (long)blockIdx.x * blockDim.x + threadIdx.x;  // over MTOK*DINNER
  if (i >= (long)MTOK * DINNER) return;
  const int c = (int)(i & (DINNER - 1));
  const long t = i >> 11;
  y16[i] = (__bf16)(hs[t * 16 + (c & 15)] + xr[t * (2 * DINNER) + DINNER + c]);
}

// --------------------------------------------------------------- launch ----
static inline char* bump(char*& p, size_t bytes) {
  char* r = p;
  p += (bytes + 255) & ~(size_t)255;
  return r;
}

extern "C" void kernel_launch(void* const* d_in, const int* in_sizes, int n_in,
                              void* d_out, int out_size, void* d_ws, size_t ws_size,
                              hipStream_t stream) {
  const float* x    = (const float*)d_in[0];
  const int*   msk  = (const int*)d_in[1];
  const float* Wq   = (const float*)d_in[2];  const float* bq   = (const float*)d_in[3];
  const float* Wk   = (const float*)d_in[4];  const float* bk   = (const float*)d_in[5];
  const float* Wv   = (const float*)d_in[6];  const float* bv   = (const float*)d_in[7];
  const float* Wo   = (const float*)d_in[8];  const float* bo   = (const float*)d_in[9];
  const float* inW  = (const float*)d_in[10]; const float* inb  = (const float*)d_in[11];
  const float* cw   = (const float*)d_in[12]; const float* cb   = (const float*)d_in[13];
  const float* xpW  = (const float*)d_in[14]; const float* xpb  = (const float*)d_in[15];
  const float* dtp  = (const float*)d_in[16]; const float* Amat = (const float*)d_in[17];
  const float* outW = (const float*)d_in[19]; const float* outb = (const float*)d_in[20];
  const float* f1W  = (const float*)d_in[21]; const float* f1b  = (const float*)d_in[22];
  const float* f2W  = (const float*)d_in[23]; const float* f2b  = (const float*)d_in[24];
  const float* ln1g = (const float*)d_in[25]; const float* ln1b = (const float*)d_in[26];
  const float* ln2g = (const float*)d_in[27]; const float* ln2b = (const float*)d_in[28];
  const float* ln3g = (const float*)d_in[29]; const float* ln3b = (const float*)d_in[30];
  float* outp = (float*)d_out;
  (void)in_sizes; (void)n_in; (void)out_size; (void)ws_size;

  // ---- workspace carve-up (aliased serially) ----
  char* p = (char*)d_ws;
  float* xr_f   = (float*)bump(p, (size_t)MTOK * 4096 * 4);    // xr / ffn hidden
  float* q_f    = (float*)bump(p, (size_t)MTOK * DMODEL * 4);  // q / attn-out
  float* k_f    = (float*)bump(p, (size_t)MTOK * DMODEL * 4);  // k / o / ffn-out
  float* v_f    = (float*)bump(p, (size_t)MTOK * DMODEL * 4);  // v / mamba-out
  float* x1_f   = (float*)bump(p, (size_t)MTOK * DMODEL * 4);
  float* x2_f   = (float*)bump(p, (size_t)MTOK * DMODEL * 4);
  float* xdbl_f = (float*)bump(p, (size_t)MTOK * 32 * 4);
  float* hs_f   = (float*)bump(p, (size_t)MTOK * 16 * 4);

  __bf16* wq16  = (__bf16*)bump(p, (size_t)DMODEL * DMODEL * 2);
  __bf16* wk16  = (__bf16*)bump(p, (size_t)DMODEL * DMODEL * 2);
  __bf16* wv16  = (__bf16*)bump(p, (size_t)DMODEL * DMODEL * 2);
  __bf16* wo16  = (__bf16*)bump(p, (size_t)DMODEL * DMODEL * 2);
  __bf16* inW16 = (__bf16*)bump(p, (size_t)(2 * DINNER) * DMODEL * 2);
  __bf16* xpW16 = (__bf16*)bump(p, (size_t)(2 * DSTATE) * DINNER * 2);
  __bf16* outW16= (__bf16*)bump(p, (size_t)DMODEL * DINNER * 2);
  __bf16* f1W16 = (__bf16*)bump(p, (size_t)DFF * DMODEL * 2);
  __bf16* f2W16 = (__bf16*)bump(p, (size_t)DMODEL * DFF * 2);
  __bf16* xg16  = (__bf16*)bump(p, (size_t)MTOK * DMODEL * 2);  // x/attn/x1/x2
  __bf16* big16 = (__bf16*)bump(p, (size_t)MTOK * 4096 * 2);    // 16M halfs
  __bf16* q16   = big16;
  __bf16* k16   = big16 + (size_t)MTOK * DMODEL;
  __bf16* vt16  = big16 + (size_t)MTOK * 2 * DMODEL;
  __bf16* xi216 = big16;                                 // 8M halfs (reuse)
  __bf16* y16   = big16 + (size_t)MTOK * DINNER;         // 8M halfs (reuse)
  __bf16* h16   = big16;                                 // 16M halfs (reuse)

  auto cvt = [&](const float* s, __bf16* d, size_t n) {
    cvt_f32_to_bf16<<<dim3((unsigned)((n + 255) / 256)), dim3(256), 0, stream>>>(
        s, d, (int)n);
  };
  auto gemm = [&](const __bf16* Ax, const __bf16* Wx, const float* bias, float* C,
                  __bf16* C16, int Mm, int Nn, int Kk, int act) {
    if (Nn % 64 == 0)
      gemm_bf16_tdm<<<dim3(Nn / 64, Mm / 64), dim3(128), 0, stream>>>(
          Ax, Wx, bias, C, C16, Mm, Nn, Kk, act);
    else
      gemm_bf16_nt<2><<<dim3(Nn / 32, Mm / 64), dim3(128), 0, stream>>>(
          Ax, Wx, bias, C, C16, Mm, Nn, Kk, act);
  };

  // ---- 0: conversions ----
  cvt(Wq, wq16, (size_t)DMODEL * DMODEL);
  cvt(Wk, wk16, (size_t)DMODEL * DMODEL);
  cvt(Wv, wv16, (size_t)DMODEL * DMODEL);
  cvt(Wo, wo16, (size_t)DMODEL * DMODEL);
  cvt(inW, inW16, (size_t)(2 * DINNER) * DMODEL);
  cvt(xpW, xpW16, (size_t)(2 * DSTATE) * DINNER);
  cvt(outW, outW16, (size_t)DMODEL * DINNER);
  cvt(f1W, f1W16, (size_t)DFF * DMODEL);
  cvt(f2W, f2W16, (size_t)DMODEL * DFF);
  cvt(x, xg16, (size_t)MTOK * DMODEL);

  // ---- 1: attention ----
  gemm(xg16, wq16, bq, q_f, q16, MTOK, DMODEL, DMODEL, 0);
  gemm(xg16, wk16, bk, k_f, k16, MTOK, DMODEL, DMODEL, 0);
  gemm(xg16, wv16, bv, v_f, nullptr, MTOK, DMODEL, DMODEL, 0);
  transpose_v<<<dim3(MTOK * DMODEL / 256), dim3(256), 0, stream>>>(v_f, vt16);
  flash_attn<<<dim3(SEQ / 64, NHEADS, NBATCH), dim3(128), 0, stream>>>(
      q16, k16, vt16, msk, q_f);
  cvt(q_f, xg16, (size_t)MTOK * DMODEL);
  gemm(xg16, wo16, bo, k_f, nullptr, MTOK, DMODEL, DMODEL, 0);
  add_ln<<<dim3(MTOK), dim3(256), 0, stream>>>(x, k_f, ln1g, ln1b, x1_f, xg16);

  // ---- 2: mamba ----
  gemm(xg16, inW16, inb, xr_f, nullptr, MTOK, 2 * DINNER, DMODEL, 0);
  conv_silu<<<dim3(MTOK * DINNER / 256), dim3(256), 0, stream>>>(xr_f, cw, cb, xi216);
  gemm(xi216, xpW16, xpb, xdbl_f, nullptr, MTOK, 2 * DSTATE, DINNER, 0);
  mamba_scan<<<dim3(1), dim3(64), 0, stream>>>(xdbl_f, dtp, Amat, hs_f);
  tile_add_res<<<dim3(MTOK * DINNER / 256), dim3(256), 0, stream>>>(hs_f, xr_f, y16);
  gemm(y16, outW16, outb, v_f, nullptr, MTOK, DMODEL, DINNER, 0);
  add_ln<<<dim3(MTOK), dim3(256), 0, stream>>>(x1_f, v_f, ln2g, ln2b, x2_f, xg16);

  // ---- 3: FFN ----
  gemm(xg16, f1W16, f1b, xr_f, h16, MTOK, DFF, DMODEL, 1);
  gemm(h16, f2W16, f2b, k_f, nullptr, MTOK, DMODEL, DFF, 0);
  add_ln<<<dim3(MTOK), dim3(256), 0, stream>>>(x2_f, k_f, ln3g, ln3b, outp, nullptr);
}
